// SpikingNet_33268816675085
// MI455X (gfx1250) — matrix operand order repaired and checked by reading, not compiled, not run on hardware
//
#include <hip/hip_runtime.h>

#define NUM_STEPS 100
#define BATCH 256
#define NUM_IN 1024
#define NUM_HID 2048
#define NUM_OUT 512
#define ALPHA 0.9f
#define BETA 0.9f

#define KPANEL 256          // K-halves staged per LDS panel
#define APITCH 264          // row pitch (halves): 16B/row bank skew, 16B aligned

typedef _Float16 v16h __attribute__((ext_vector_type(16)));
typedef _Float16 v4h  __attribute__((ext_vector_type(4)));
typedef float    v8f  __attribute__((ext_vector_type(8)));

union Frag { v16h v; uint4 q[2]; };

// B fragment given a precomputed column base pointer (pb = W + col*ld + half*16).
// Inner-loop k offset is an immediate-friendly element offset.
__device__ __forceinline__ v16h load_b_at(const _Float16* __restrict__ pb, int kofs) {
  Frag f;
  f.q[0] = *(const uint4*)(pb + kofs);
  f.q[1] = *(const uint4*)(pb + kofs + 8);
  return f.v;
}

// A fragment from an LDS panel (rows relative to panel).
__device__ __forceinline__ v16h load_a_lds(const _Float16* __restrict__ pa, int kofs) {
  Frag f;
  f.q[0] = *(const uint4*)(pa + kofs);
  f.q[1] = *(const uint4*)(pa + kofs + 8);
  return f.v;
}

// A fragment from f32 row-major source (pa = src + row*ld + half*16), f16 convert.
__device__ __forceinline__ v16h load_a_f32_at(const float* __restrict__ pa, int kofs) {
  float4 f0 = *(const float4*)(pa + kofs);
  float4 f1 = *(const float4*)(pa + kofs + 4);
  float4 f2 = *(const float4*)(pa + kofs + 8);
  float4 f3 = *(const float4*)(pa + kofs + 12);
  v16h a;
  a[0]  = (_Float16)f0.x; a[1]  = (_Float16)f0.y; a[2]  = (_Float16)f0.z; a[3]  = (_Float16)f0.w;
  a[4]  = (_Float16)f1.x; a[5]  = (_Float16)f1.y; a[6]  = (_Float16)f1.z; a[7]  = (_Float16)f1.w;
  a[8]  = (_Float16)f2.x; a[9]  = (_Float16)f2.y; a[10] = (_Float16)f2.z; a[11] = (_Float16)f2.w;
  a[12] = (_Float16)f3.x; a[13] = (_Float16)f3.y; a[14] = (_Float16)f3.z; a[15] = (_Float16)f3.w;
  return a;
}

// Async global->LDS stage of a 16 x KPANEL f16 panel (8 KB), 128 threads.
// Tracked by ASYNCcnt; caller pairs with s_wait_asynccnt + barrier.
__device__ __forceinline__ void stage_panel_async(
    const _Float16* __restrict__ src, int rowbase, int ld, int kpanel,
    _Float16 (*tile)[APITCH]) {
  const int t = threadIdx.x;
#pragma unroll
  for (int pass = 0; pass < (16 * KPANEL) / (8 * 128); ++pass) {   // 4 passes
    int c   = pass * 128 + t;          // 0..511 b128 chunks
    int row = c >> 5;                  // 32 chunks per row
    int col = (c & 31) * 8;            // halves
    unsigned long long g =
        (unsigned long long)(uintptr_t)(src + (size_t)(rowbase + row) * ld + kpanel + col);
    unsigned lds = (unsigned)(uintptr_t)(&tile[row][col]);
    asm volatile("global_load_async_to_lds_b128 %0, %1, off"
                 :: "v"(lds), "v"(g) : "memory");
  }
}

#define WMMA(a, b, c) __builtin_amdgcn_wmma_f32_16x16x32_f16(false, (a), false, (b), \
                                                             (short)0, (c), false, false)

// ---------------- Kernel 1: hidden layer (RSynaptic) ----------------
// acc = x_t @ w1^T + spk_prev @ wrec^T, then fused synaptic/membrane update.
__global__ void __launch_bounds__(128) snn_hidden(
    const float*    __restrict__ xt,       // [BATCH, NUM_IN] fp32
    const _Float16* __restrict__ w1h,      // [NUM_HID, NUM_IN]
    const _Float16* __restrict__ wrech,    // [NUM_HID, NUM_HID]
    const _Float16* __restrict__ spkPrev,  // [BATCH, NUM_HID]
    _Float16*       __restrict__ spkNext,  // [BATCH, NUM_HID]
    float*          __restrict__ syn1,     // [BATCH, NUM_HID]
    float*          __restrict__ mem1)     // [BATCH, NUM_HID]
{
  __shared__ __align__(16) _Float16 atile[2][16][APITCH];

  const int lane = threadIdx.x & 31;
  const int wib  = threadIdx.x >> 5;
  const int wid  = blockIdx.x * 4 + wib;
  const int NT   = NUM_HID / 64;              // 32 N-tiles of 64
  const int mbase = (wid / NT) * 16;          // identical for all 4 waves in block
  const int nbase = (wid % NT) * 64;
  const int half = lane >> 4;
  const int lr   = lane & 15;
  const int arow = mbase + lr;

  // Kick off async stage of spike panel 0; it hides behind phase 1.
  stage_panel_async(spkPrev, mbase, NUM_HID, 0, atile[0]);

  v8f c[4] = {};

  // Phase 1: x_t @ w1^T  (K = 1024, fp32 A converted in-register).
  {
    const float*    pa  = xt  + (size_t)arow * NUM_IN + half * 16;
    const _Float16* pb0 = w1h + (size_t)(nbase +  0 + lr) * NUM_IN + half * 16;
    const _Float16* pb1 = w1h + (size_t)(nbase + 16 + lr) * NUM_IN + half * 16;
    const _Float16* pb2 = w1h + (size_t)(nbase + 32 + lr) * NUM_IN + half * 16;
    const _Float16* pb3 = w1h + (size_t)(nbase + 48 + lr) * NUM_IN + half * 16;
#pragma unroll 2
    for (int k = 0; k < NUM_IN; k += 32) {
      v16h a  = load_a_f32_at(pa, k);
      v16h b0 = load_b_at(pb0, k);
      v16h b1 = load_b_at(pb1, k);
      v16h b2 = load_b_at(pb2, k);
      v16h b3 = load_b_at(pb3, k);
      c[0] = WMMA(a, b0, c[0]);
      c[1] = WMMA(a, b1, c[1]);
      c[2] = WMMA(a, b2, c[2]);
      c[3] = WMMA(a, b3, c[3]);
    }
  }

  // Phase 2: spk_prev @ wrec^T  (K = 2048), A via async-LDS double buffer.
  {
    const _Float16* pb0 = wrech + (size_t)(nbase +  0 + lr) * NUM_HID + half * 16;
    const _Float16* pb1 = wrech + (size_t)(nbase + 16 + lr) * NUM_HID + half * 16;
    const _Float16* pb2 = wrech + (size_t)(nbase + 32 + lr) * NUM_HID + half * 16;
    const _Float16* pb3 = wrech + (size_t)(nbase + 48 + lr) * NUM_HID + half * 16;
    const int NPAN = NUM_HID / KPANEL;        // 8 panels
    for (int p = 0; p < NPAN; ++p) {
      asm volatile("s_wait_asynccnt 0x0" ::: "memory");
      __syncthreads();
      if (p + 1 < NPAN)
        stage_panel_async(spkPrev, mbase, NUM_HID, (p + 1) * KPANEL, atile[(p + 1) & 1]);
      const _Float16* pa = &atile[p & 1][lr][half * 16];
      const int kg = p * KPANEL;
#pragma unroll
      for (int kk = 0; kk < KPANEL; kk += 32) {
        v16h a  = load_a_lds(pa, kk);
        v16h b0 = load_b_at(pb0, kg + kk);
        v16h b1 = load_b_at(pb1, kg + kk);
        v16h b2 = load_b_at(pb2, kg + kk);
        v16h b3 = load_b_at(pb3, kg + kk);
        c[0] = WMMA(a, b0, c[0]);
        c[1] = WMMA(a, b1, c[1]);
        c[2] = WMMA(a, b2, c[2]);
        c[3] = WMMA(a, b3, c[3]);
      }
    }
  }

  // Fused RSynaptic update (subtract reset, reset_delay=False)
#pragma unroll
  for (int j = 0; j < 4; ++j) {
    const int col = nbase + 16 * j + lr;
#pragma unroll
    for (int r = 0; r < 8; ++r) {
      const int row = mbase + r + 8 * half;
      const int idx = row * NUM_HID + col;
      float acc     = c[j][r];
      float syn_new = ALPHA * syn1[idx] + acc;
      float mo      = mem1[idx];
      float reset   = (mo > 1.0f) ? 1.0f : 0.0f;
      float m       = BETA * mo + syn_new - reset;
      float spk     = (m > 1.0f) ? 1.0f : 0.0f;
      m             = m - spk + reset;          // immediate (spk - reset) subtract
      syn1[idx]     = syn_new;
      mem1[idx]     = m;
      spkNext[idx]  = (_Float16)spk;
    }
  }
}

// ---------------- Kernel 2: output layer (Leaky) ----------------
__global__ void __launch_bounds__(128) snn_out(
    const _Float16* __restrict__ spk,   // [BATCH, NUM_HID]
    const _Float16* __restrict__ w2h,   // [NUM_OUT, NUM_HID]
    float*          __restrict__ mem2,  // [BATCH, NUM_OUT]
    float*          __restrict__ out,   // d_out
    int t)
{
  __shared__ __align__(16) _Float16 atile[2][16][APITCH];

  const int lane = threadIdx.x & 31;
  const int wib  = threadIdx.x >> 5;
  const int wid  = blockIdx.x * 4 + wib;
  const int NT   = NUM_OUT / 64;              // 8 N-tiles
  const int mbase = (wid / NT) * 16;          // identical for all 4 waves in block
  const int nbase = (wid % NT) * 64;
  const int half = lane >> 4;
  const int lr   = lane & 15;

  stage_panel_async(spk, mbase, NUM_HID, 0, atile[0]);

  v8f c[4] = {};

  {
    const _Float16* pb0 = w2h + (size_t)(nbase +  0 + lr) * NUM_HID + half * 16;
    const _Float16* pb1 = w2h + (size_t)(nbase + 16 + lr) * NUM_HID + half * 16;
    const _Float16* pb2 = w2h + (size_t)(nbase + 32 + lr) * NUM_HID + half * 16;
    const _Float16* pb3 = w2h + (size_t)(nbase + 48 + lr) * NUM_HID + half * 16;
    const int NPAN = NUM_HID / KPANEL;        // 8 panels
    for (int p = 0; p < NPAN; ++p) {
      asm volatile("s_wait_asynccnt 0x0" ::: "memory");
      __syncthreads();
      if (p + 1 < NPAN)
        stage_panel_async(spk, mbase, NUM_HID, (p + 1) * KPANEL, atile[(p + 1) & 1]);
      const _Float16* pa = &atile[p & 1][lr][half * 16];
      const int kg = p * KPANEL;
#pragma unroll
      for (int kk = 0; kk < KPANEL; kk += 32) {
        v16h a  = load_a_lds(pa, kk);
        v16h b0 = load_b_at(pb0, kg + kk);
        v16h b1 = load_b_at(pb1, kg + kk);
        v16h b2 = load_b_at(pb2, kg + kk);
        v16h b3 = load_b_at(pb3, kg + kk);
        c[0] = WMMA(a, b0, c[0]);
        c[1] = WMMA(a, b1, c[1]);
        c[2] = WMMA(a, b2, c[2]);
        c[3] = WMMA(a, b3, c[3]);
      }
    }
  }

  const size_t step_off = (size_t)t * BATCH * NUM_OUT;
  const size_t mem_base = (size_t)NUM_STEPS * BATCH * NUM_OUT;
#pragma unroll
  for (int j = 0; j < 4; ++j) {
    const int col = nbase + 16 * j + lr;
#pragma unroll
    for (int r = 0; r < 8; ++r) {
      const int row = mbase + r + 8 * half;
      const int idx = row * NUM_OUT + col;
      float mo    = mem2[idx];
      float reset = (mo > 1.0f) ? 1.0f : 0.0f;
      float m     = BETA * mo + c[j][r] - reset;
      float s     = (m > 1.0f) ? 1.0f : 0.0f;   // reset_delay=True: applied next step
      mem2[idx] = m;
      out[step_off + idx]            = s;
      out[mem_base + step_off + idx] = m;
    }
  }
}

// ---------------- Prep kernels ----------------
__global__ void cvt_f32_to_f16_x4(const float* __restrict__ s,
                                  _Float16* __restrict__ d, int n4) {
  int i = blockIdx.x * blockDim.x + threadIdx.x;
  if (i < n4) {
    float4 f = ((const float4*)s)[i];
    v4h h;
    h[0] = (_Float16)f.x; h[1] = (_Float16)f.y;
    h[2] = (_Float16)f.z; h[3] = (_Float16)f.w;
    ((v4h*)d)[i] = h;
  }
}

__global__ void zero_f32(float* __restrict__ p, int n) {
  int i = blockIdx.x * blockDim.x + threadIdx.x;
  if (i < n) p[i] = 0.0f;
}

__global__ void zero_f16(_Float16* __restrict__ p, int n) {
  int i = blockIdx.x * blockDim.x + threadIdx.x;
  if (i < n) p[i] = (_Float16)0.0f;
}

extern "C" void kernel_launch(void* const* d_in, const int* in_sizes, int n_in,
                              void* d_out, int out_size, void* d_ws, size_t ws_size,
                              hipStream_t stream) {
  const float* x    = (const float*)d_in[0];   // [100,256,1024]
  const float* w1   = (const float*)d_in[1];   // [2048,1024]
  const float* wrec = (const float*)d_in[2];   // [2048,2048]
  const float* w2   = (const float*)d_in[3];   // [512,2048]
  float* out = (float*)d_out;

  char* ws = (char*)d_ws;
  const size_t MB = 1024 * 1024;
  _Float16* w1h   = (_Float16*)(ws);                 //  4 MB
  _Float16* wrech = (_Float16*)(ws + 4 * MB);        //  8 MB
  _Float16* w2h   = (_Float16*)(ws + 12 * MB);       //  2 MB
  _Float16* spkA  = (_Float16*)(ws + 14 * MB);       //  1 MB
  _Float16* spkB  = (_Float16*)(ws + 15 * MB);       //  1 MB
  float*    syn1  = (float*)   (ws + 16 * MB);       //  2 MB
  float*    mem1  = (float*)   (ws + 18 * MB);       //  2 MB
  float*    mem2  = (float*)   (ws + 20 * MB);       // 0.5 MB

  // Convert weights to f16 once per call.
  {
    int n4 = NUM_HID * NUM_IN / 4;
    cvt_f32_to_f16_x4<<<(n4 + 255) / 256, 256, 0, stream>>>(w1, w1h, n4);
    n4 = NUM_HID * NUM_HID / 4;
    cvt_f32_to_f16_x4<<<(n4 + 255) / 256, 256, 0, stream>>>(wrec, wrech, n4);
    n4 = NUM_OUT * NUM_HID / 4;
    cvt_f32_to_f16_x4<<<(n4 + 255) / 256, 256, 0, stream>>>(w2, w2h, n4);
  }
  // Zero state (syn1 | mem1 | mem2 are contiguous).
  {
    int nf = 2 * BATCH * NUM_HID + BATCH * NUM_OUT;
    zero_f32<<<(nf + 255) / 256, 256, 0, stream>>>(syn1, nf);
    int nh = BATCH * NUM_HID;
    zero_f16<<<(nh + 255) / 256, 256, 0, stream>>>(spkA, nh);
  }

  // 100 sequential steps, ping-pong spike buffers.
  _Float16* sp = spkA;
  _Float16* sn = spkB;
  const int blocks1 = (BATCH / 16) * (NUM_HID / 64) / 4;  // 128
  const int blocks2 = (BATCH / 16) * (NUM_OUT / 64) / 4;  // 32
  for (int t = 0; t < NUM_STEPS; ++t) {
    snn_hidden<<<blocks1, 128, 0, stream>>>(x + (size_t)t * BATCH * NUM_IN,
                                            w1h, wrech, sp, sn, syn1, mem1);
    snn_out<<<blocks2, 128, 0, stream>>>(sn, w2h, mem2, out, t);
    _Float16* tmp = sp; sp = sn; sn = tmp;
  }
}
